// TransformerEncoderLayer_11046655885679
// MI455X (gfx1250) — compile-verified, matrix operand order checked by reading, not run on hardware
//
#include <hip/hip_runtime.h>
#include <hip/hip_bf16.h>
#include <math.h>

typedef __bf16 bf16;
typedef __attribute__((ext_vector_type(8)))  __bf16 v8bf;
typedef __attribute__((ext_vector_type(16))) __bf16 v16bf;
typedef __attribute__((ext_vector_type(8)))  float  v8f;
typedef __attribute__((ext_vector_type(4)))  unsigned int u32x4;
typedef __attribute__((ext_vector_type(8)))  int i32x8;
typedef __attribute__((ext_vector_type(4)))  int i32x4;

union FragBF { v16bf v; v8bf h[2]; };

static __device__ __forceinline__ v8f vzero8() {
    v8f z;
#pragma unroll
    for (int i = 0; i < 8; ++i) z[i] = 0.0f;
    return z;
}

static __device__ __forceinline__ v8f wmma_bf16(const FragBF& a, const FragBF& b, v8f c) {
    // D = A(16x32 bf16) * B(32x16 bf16) + C(16x16 f32)
    return __builtin_amdgcn_wmma_f32_16x16x32_bf16(false, a.v, false, b.v,
                                                   (short)0, c, false, false);
}

// ---------------------------------------------------------------------------
// DPP row reductions within a 16-lane DPP row (no LDS, no waits).
// ---------------------------------------------------------------------------
template <int CTRL>
static __device__ __forceinline__ float dpp_f32(float v) {
    int r = __builtin_amdgcn_update_dpp(0, __float_as_int(v), CTRL, 0xF, 0xF, true);
    return __int_as_float(r);
}
static __device__ __forceinline__ float rowmax16(float v) {
    v = fmaxf(v, dpp_f32<0xB1>(v));    // quad_perm [1,0,3,2]
    v = fmaxf(v, dpp_f32<0x4E>(v));    // quad_perm [2,3,0,1]
    v = fmaxf(v, dpp_f32<0x141>(v));   // row_half_mirror
    v = fmaxf(v, dpp_f32<0x140>(v));   // row_mirror
    return v;
}
static __device__ __forceinline__ float rowsum16(float v) {
    v += dpp_f32<0xB1>(v);
    v += dpp_f32<0x4E>(v);
    v += dpp_f32<0x141>(v);
    v += dpp_f32<0x140>(v);
    return v;
}

// ---------------------------------------------------------------------------
// Tensor Data Mover: async 2D bf16 tile load Global -> LDS with HW padding.
//   pad codes per ISA D# group1: interval 3 -> every 16 DWORDs (32 bf16 row),
//   interval 4 -> every 32 DWORDs (64 bf16 row); amount 3 -> 4 DWORDs (8 bf16).
// ---------------------------------------------------------------------------
static __device__ __forceinline__ void tdm_load_2d_bf16(
    const bf16* gsrc, bf16* lds_dst,
    unsigned tile_w, unsigned tile_h, unsigned row_stride,
    unsigned pad_interval_code, unsigned pad_amount_code)
{
    unsigned long long ga = (unsigned long long)(uintptr_t)gsrc;
    u32x4 g0;
    g0.x = 1u;                                            // count=1 (user D#)
    g0.y = (unsigned)(uintptr_t)lds_dst;                  // lds_addr (bytes)
    g0.z = (unsigned)ga;                                  // global_addr[31:0]
    g0.w = (unsigned)((ga >> 32) & 0x01ffffffu) | (2u << 30);  // addr[56:32], type=2
    i32x8 g1;
    g1[0] = (int)((1u << 16)                              // data_size = 2 bytes
                 | (1u << 20)                             // pad_enable
                 | (pad_interval_code << 22)
                 | (pad_amount_code << 25));
    g1[1] = (int)((tile_w & 0xffffu) << 16);              // tensor_dim0[15:0]
    g1[2] = (int)((tile_h & 0xffffu) << 16);              // tensor_dim1[15:0]
    g1[3] = (int)((tile_w & 0xffffu) << 16);              // tile_dim0
    g1[4] = (int)(tile_h & 0xffffu);                      // tile_dim1 (tile_dim2=0)
    g1[5] = (int)row_stride;                              // tensor_dim0_stride[31:0]
    g1[6] = 0;                                            // stride hi / dim1_stride
    g1[7] = 0;
    i32x4 z4; z4[0] = 0; z4[1] = 0; z4[2] = 0; z4[3] = 0;
    i32x8 z8;
#pragma unroll
    for (int i = 0; i < 8; ++i) z8[i] = 0;
    __builtin_amdgcn_tensor_load_to_lds(g0, g1, z4, z4, z8, 0);
}

// ---------------------------------------------------------------------------
// fp32 -> bf16 elementwise convert
// ---------------------------------------------------------------------------
__global__ __launch_bounds__(256) void cvt_f32_bf16(const float* __restrict__ in,
                                                    bf16* __restrict__ out, int n) {
    int i = blockIdx.x * 256 + threadIdx.x;
    if (i < n) out[i] = (bf16)in[i];
}

// ---------------------------------------------------------------------------
// LayerNorm over D=1024, one row per block (256 threads), bf16 output
// ---------------------------------------------------------------------------
__global__ __launch_bounds__(256) void ln_kernel(const float* __restrict__ x,
                                                 const float* __restrict__ g,
                                                 const float* __restrict__ bb,
                                                 bf16* __restrict__ out) {
    const int Dm = 1024;
    int row = blockIdx.x;
    int tid = threadIdx.x, lane = tid & 31, wid = tid >> 5;
    const float* xr = x + (size_t)row * Dm;
    float vals[4];
    float s = 0.f, s2 = 0.f;
#pragma unroll
    for (int i = 0; i < 4; ++i) {
        float v = xr[tid + i * 256];
        vals[i] = v; s += v; s2 += v * v;
    }
#pragma unroll
    for (int off = 16; off >= 1; off >>= 1) {
        s  += __shfl_xor(s,  off, 32);
        s2 += __shfl_xor(s2, off, 32);
    }
    __shared__ float rs[8], rs2[8];
    if (lane == 0) { rs[wid] = s; rs2[wid] = s2; }
    __syncthreads();
    s = 0.f; s2 = 0.f;
#pragma unroll
    for (int w = 0; w < 8; ++w) { s += rs[w]; s2 += rs2[w]; }
    float mu = s * (1.0f / Dm);
    float var = s2 * (1.0f / Dm) - mu * mu;
    float rstd = rsqrtf(var + 1e-5f);
    bf16* orow = out + (size_t)row * Dm;
#pragma unroll
    for (int i = 0; i < 4; ++i) {
        int c = tid + i * 256;
        orow[c] = (bf16)((vals[i] - mu) * rstd * g[c] + bb[c]);
    }
}

// ---------------------------------------------------------------------------
// WMMA GEMM: C[M,N] = A[M,K](bf16) * B[K,N](bf16) + bias (+epilogue)
//  workgroup tile 256(M)x128(N), 8 waves (4 along M x 2 along N),
//  wave tile 64x64 = 4x4 WMMA tiles -> 8 frag loads feed 16 WMMAs,
//  each fragment reused 4x from registers.
//  Double-buffered LDS; A tiles (256x32) via async TDM, B staged transposed;
//  K-loop unrolled 2x so buffer parity is compile-time.
//  EPI: 0 = bias -> bf16 ; 1 = bias+gelu -> bf16 ; 2 = bias+residual -> f32
// ---------------------------------------------------------------------------
template <int EPI>
__global__ __launch_bounds__(256) void gemm_bf16_k(const bf16* __restrict__ A,
                                                   const bf16* __restrict__ B,
                                                   const float* __restrict__ bias,
                                                   const float* __restrict__ resid,
                                                   void* __restrict__ Cout,
                                                   int M, int N, int K) {
    constexpr int LDT = 40;                 // 32 K-elems padded to 40 (80B rows)
    __shared__ bf16 As[2][256 * LDT];       // [m][k]  (TDM-padded)   40KB
    __shared__ bf16 Bt[2][128 * LDT];       // [n][k]  (transposed)   20KB

    const int tid  = threadIdx.x;
    const int lane = tid & 31;
    const int wid  = tid >> 5;
    const int lrow = lane & 15;
    const int kb8  = (lane < 16) ? 0 : 8;   // fragment K-base per half-wave
    const int rsel = (lane < 16) ? 0 : 8;   // C fragment row-base per half-wave
    const int waveM = (wid >> 1) * 64;      // 4 waves along M
    const int waveN = (wid & 1) * 64;       // 2 waves along N
    const int m0 = blockIdx.y * 256;
    const int n0 = blockIdx.x * 128;

    v8f acc[4][4];
#pragma unroll
    for (int mt = 0; mt < 4; ++mt)
#pragma unroll
        for (int nt = 0; nt < 4; ++nt) acc[mt][nt] = vzero8();

    // ---- prologue: stage k=0 into buffer 0 ----
    if (wid == 0)
        tdm_load_2d_bf16(A + (size_t)m0 * K, As[0], 32, 256, K, 3, 3);
#pragma unroll
    for (int t = 0; t < 2; ++t) {
        int i = tid + t * 256;
        int br = i >> 4, bc = (i & 15) << 3;
        v8bf bv = *(const v8bf*)(B + (size_t)br * N + (n0 + bc));
#pragma unroll
        for (int j = 0; j < 8; ++j) Bt[0][(bc + j) * LDT + br] = bv[j];
    }
    __builtin_amdgcn_s_wait_tensorcnt(0);
    __syncthreads();

    for (int k0 = 0; k0 < K; k0 += 64) {
#pragma unroll
        for (int u = 0; u < 2; ++u) {       // buffer parity is compile-time
            const int kk  = k0 + u * 32;
            const int cur = u;
            const int nxt = u ^ 1;
            const bool more = (kk + 32) < K;

            // issue next tile's DMA + global loads before computing on current
            v8bf breg[2];
            if (more) {
                if (wid == 0)
                    tdm_load_2d_bf16(A + (size_t)m0 * K + (kk + 32), As[nxt],
                                     32, 256, K, 3, 3);
#pragma unroll
                for (int t = 0; t < 2; ++t) {
                    int i = tid + t * 256;
                    int br = i >> 4, bc = (i & 15) << 3;
                    breg[t] = *(const v8bf*)(B + (size_t)(kk + 32 + br) * N +
                                             (n0 + bc));
                }
            }

            // compute on current buffer: 8 frag loads -> 16 WMMAs
            FragBF af[4], bfr[4];
#pragma unroll
            for (int mt = 0; mt < 4; ++mt) {
                const bf16* p = &As[cur][(waveM + mt * 16 + lrow) * LDT + kb8];
                af[mt].h[0] = *(const v8bf*)p;
                af[mt].h[1] = *(const v8bf*)(p + 16);
            }
#pragma unroll
            for (int nt = 0; nt < 4; ++nt) {
                const bf16* p = &Bt[cur][(waveN + nt * 16 + lrow) * LDT + kb8];
                bfr[nt].h[0] = *(const v8bf*)p;
                bfr[nt].h[1] = *(const v8bf*)(p + 16);
            }
#pragma unroll
            for (int mt = 0; mt < 4; ++mt)
#pragma unroll
                for (int nt = 0; nt < 4; ++nt)
                    acc[mt][nt] = wmma_bf16(af[mt], bfr[nt], acc[mt][nt]);

            // commit staged B registers to the other buffer
            if (more) {
#pragma unroll
                for (int t = 0; t < 2; ++t) {
                    int i = tid + t * 256;
                    int br = i >> 4, bc = (i & 15) << 3;
#pragma unroll
                    for (int j = 0; j < 8; ++j)
                        Bt[nxt][(bc + j) * LDT + br] = breg[t][j];
                }
            }
            __builtin_amdgcn_s_wait_tensorcnt(0);
            __syncthreads();
        }
    }

    // epilogue
#pragma unroll
    for (int nt = 0; nt < 4; ++nt) {
        int col = n0 + waveN + nt * 16 + lrow;
        float bv = bias[col];
#pragma unroll
        for (int mt = 0; mt < 4; ++mt) {
            int rbase = m0 + waveM + mt * 16 + rsel;
#pragma unroll
            for (int j = 0; j < 8; ++j) {
                int row = rbase + j;
                float v = acc[mt][nt][j] + bv;
                if (EPI == 1) v = 0.5f * v * (1.0f + erff(v * 0.70710678118f));
                if (EPI == 2) {
                    v += resid[(size_t)row * N + col];
                    ((float*)Cout)[(size_t)row * N + col] = v;
                } else {
                    ((bf16*)Cout)[(size_t)row * N + col] = (bf16)v;
                }
            }
        }
    }
}

// ---------------------------------------------------------------------------
// Flash attention: one block per (batch, head, 64-row q block); 128 threads.
// qkv layout [B*N, 3*D] bf16; out [B*N, D] bf16 at column h*64.
// Q/K tiles via TDM; K/V double-buffered; DPP row reductions for softmax.
// ---------------------------------------------------------------------------
__global__ __launch_bounds__(128) void attn_kernel(const bf16* __restrict__ qkv,
                                                   bf16* __restrict__ outb) {
    const int NSEQ = 2048, D3 = 3072, HD = 64, DD = 1024;
    const int qb = blockIdx.x & 31;
    const int h  = (blockIdx.x >> 5) & 15;
    const int b  = blockIdx.x >> 9;
    const int tid = threadIdx.x, lane = tid & 31, wid = tid >> 5;
    const int lrow = lane & 15;
    const int kb8  = (lane < 16) ? 0 : 8;
    const int rsel = (lane < 16) ? 0 : 8;
    const size_t bN   = (size_t)b * NSEQ;
    const size_t tok0 = bN + (size_t)qb * 64;
    const int qoff = h * HD, koff = DD + h * HD, voff = 2 * DD + h * HD;

    __shared__ bf16 Ql[64 * 72];            // Q block  [qrow][hd] (TDM-padded)
    __shared__ bf16 Kl[2][32 * 72];         // K blocks [key][hd]  (TDM-padded)
    __shared__ bf16 Vt[2][64 * 40];         // V blocks transposed [hd][key]
    __shared__ bf16 Pl[4 * 16 * 40];        // per-wave P tile [qrow][key]

    // ---- prologue: Q + first K/V block into buffer 0 ----
    if (wid == 0) {
        tdm_load_2d_bf16(qkv + tok0 * D3 + qoff, Ql, 64, 64, D3, 4, 3);
        tdm_load_2d_bf16(qkv + bN * D3 + koff, Kl[0], 64, 32, D3, 4, 3);
    }
#pragma unroll
    for (int t = 0; t < 2; ++t) {
        int i = tid + t * 128;
        int r = i >> 3, c = (i & 7) << 3;
        v8bf vv = *(const v8bf*)(qkv + (bN + r) * D3 + voff + c);
#pragma unroll
        for (int j = 0; j < 8; ++j) Vt[0][(c + j) * 40 + r] = vv[j];
    }
    __builtin_amdgcn_s_wait_tensorcnt(0);
    __syncthreads();

    FragBF qf[2];
#pragma unroll
    for (int s = 0; s < 2; ++s) {
        const bf16* p = &Ql[(wid * 16 + lrow) * 72 + s * 32 + kb8];
        qf[s].h[0] = *(const v8bf*)p;
        qf[s].h[1] = *(const v8bf*)(p + 16);
    }

    v8f accO[4];
#pragma unroll
    for (int nt = 0; nt < 4; ++nt) accO[nt] = vzero8();
    float mrun[8], lsum[8];
#pragma unroll
    for (int j = 0; j < 8; ++j) { mrun[j] = -3.0e38f; lsum[j] = 0.0f; }

    for (int kb = 0; kb < NSEQ / 32; ++kb) {
        const int cur = kb & 1;
        const int nxt = cur ^ 1;
        const bool more = (kb + 1) < (NSEQ / 32);

        // issue next block's DMA + V loads before computing on current
        v8bf vreg[2];
        if (more) {
            size_t ktn = bN + (size_t)(kb + 1) * 32;
            if (wid == 0)
                tdm_load_2d_bf16(qkv + ktn * D3 + koff, Kl[nxt], 64, 32, D3, 4, 3);
#pragma unroll
            for (int t = 0; t < 2; ++t) {
                int i = tid + t * 128;
                int r = i >> 3, c = (i & 7) << 3;
                vreg[t] = *(const v8bf*)(qkv + (ktn + r) * D3 + voff + c);
            }
        }

        // S = Q * K^T on current buffer
        v8f sacc[2];
#pragma unroll
        for (int nt = 0; nt < 2; ++nt) {
            sacc[nt] = vzero8();
#pragma unroll
            for (int s = 0; s < 2; ++s) {
                FragBF kf;
                const bf16* p = &Kl[cur][(nt * 16 + lrow) * 72 + s * 32 + kb8];
                kf.h[0] = *(const v8bf*)p;
                kf.h[1] = *(const v8bf*)(p + 16);
                sacc[nt] = wmma_bf16(qf[s], kf, sacc[nt]);
            }
        }

        // online softmax with DPP row reductions
        float p0[8], p1[8];
#pragma unroll
        for (int j = 0; j < 8; ++j) {
            float a = sacc[0][j] * 0.125f;   // scale = 1/sqrt(64)
            float c = sacc[1][j] * 0.125f;
            float mx = rowmax16(fmaxf(a, c));
            float mn = fmaxf(mrun[j], mx);
            float rsc = __expf(mrun[j] - mn);
            float e0 = __expf(a - mn), e1 = __expf(c - mn);
            float sm = rowsum16(e0 + e1);
            lsum[j] = lsum[j] * rsc + sm;
            mrun[j] = mn;
#pragma unroll
            for (int nt = 0; nt < 4; ++nt) accO[nt][j] *= rsc;
            p0[j] = e0; p1[j] = e1;
        }

        // stage P through per-wave LDS to re-layout C-frag -> A-frag
        bf16* Pw = &Pl[wid * 16 * 40];
#pragma unroll
        for (int j = 0; j < 8; ++j) {
            int r = j + rsel;
            Pw[r * 40 + lrow]      = (bf16)p0[j];
            Pw[r * 40 + 16 + lrow] = (bf16)p1[j];
        }
        FragBF pf;
        {
            const bf16* pp = &Pw[lrow * 40 + kb8];
            pf.h[0] = *(const v8bf*)pp;
            pf.h[1] = *(const v8bf*)(pp + 16);
        }
#pragma unroll
        for (int nt = 0; nt < 4; ++nt) {
            FragBF vf;
            const bf16* pv = &Vt[cur][(nt * 16 + lrow) * 40 + kb8];
            vf.h[0] = *(const v8bf*)pv;
            vf.h[1] = *(const v8bf*)(pv + 16);
            accO[nt] = wmma_bf16(pf, vf, accO[nt]);
        }

        // commit staged V registers to the other buffer
        if (more) {
#pragma unroll
            for (int t = 0; t < 2; ++t) {
                int i = tid + t * 128;
                int r = i >> 3, c = (i & 7) << 3;
#pragma unroll
                for (int j = 0; j < 8; ++j) Vt[nxt][(c + j) * 40 + r] = vreg[t][j];
            }
        }
        __builtin_amdgcn_s_wait_tensorcnt(0);
        __syncthreads();
    }

#pragma unroll
    for (int nt = 0; nt < 4; ++nt)
#pragma unroll
        for (int j = 0; j < 8; ++j) {
            float ov = accO[nt][j] / lsum[j];
            size_t row = tok0 + wid * 16 + j + rsel;
            outb[row * DD + h * HD + nt * 16 + lrow] = (bf16)ov;
        }
}

// ---------------------------------------------------------------------------
// Host orchestration
// ---------------------------------------------------------------------------
extern "C" void kernel_launch(void* const* d_in, const int* in_sizes, int n_in,
                              void* d_out, int out_size, void* d_ws, size_t ws_size,
                              hipStream_t stream) {
    const float* x     = (const float*)d_in[0];
    const float* ln1_g = (const float*)d_in[1];
    const float* ln1_b = (const float*)d_in[2];
    const float* w_qkv = (const float*)d_in[3];
    const float* b_qkv = (const float*)d_in[4];
    const float* w_out = (const float*)d_in[5];
    const float* b_out = (const float*)d_in[6];
    const float* ln2_g = (const float*)d_in[7];
    const float* ln2_b = (const float*)d_in[8];
    const float* w_fc1 = (const float*)d_in[9];
    const float* b_fc1 = (const float*)d_in[10];
    const float* w_fc2 = (const float*)d_in[11];
    const float* b_fc2 = (const float*)d_in[12];

    const int Dm = 1024, FF = 4096, T = 4096, D3 = 3072;

    char* p = (char*)d_ws;
    auto alloc = [&](size_t bytes) -> char* {
        char* r = p; p += (bytes + 255) & ~(size_t)255; return r;
    };
    bf16*  wqkvB = (bf16*)alloc((size_t)Dm * D3 * 2);
    bf16*  woutB = (bf16*)alloc((size_t)Dm * Dm * 2);
    bf16*  wfc1B = (bf16*)alloc((size_t)Dm * FF * 2);
    bf16*  wfc2B = (bf16*)alloc((size_t)FF * Dm * 2);
    bf16*  lnB   = (bf16*)alloc((size_t)T * Dm * 2);
    float* x1    = (float*)alloc((size_t)T * Dm * 4);
    bf16*  qkvB  = (bf16*)alloc((size_t)T * D3 * 2);
    bf16*  attnB = (bf16*)alloc((size_t)T * Dm * 2);
    bf16*  fc1B  = qkvB;   // reuse qkv+attn region (dead after out-proj)

    auto cvt = [&](const float* src, bf16* dst, int n) {
        cvt_f32_bf16<<<(n + 255) / 256, 256, 0, stream>>>(src, dst, n);
    };
    cvt(w_qkv, wqkvB, Dm * D3);
    cvt(w_out, woutB, Dm * Dm);
    cvt(w_fc1, wfc1B, Dm * FF);
    cvt(w_fc2, wfc2B, FF * Dm);

    // LN1
    ln_kernel<<<T, 256, 0, stream>>>(x, ln1_g, ln1_b, lnB);
    // QKV GEMM: [4096,1024]x[1024,3072] -> bf16
    gemm_bf16_k<0><<<dim3(D3 / 128, T / 256), 256, 0, stream>>>(
        lnB, wqkvB, b_qkv, nullptr, (void*)qkvB, T, D3, Dm);
    // attention
    attn_kernel<<<2 * 16 * 32, 128, 0, stream>>>(qkvB, attnB);
    // out-proj + residual -> f32 x1
    gemm_bf16_k<2><<<dim3(Dm / 128, T / 256), 256, 0, stream>>>(
        attnB, woutB, b_out, x, (void*)x1, T, Dm, Dm);
    // LN2
    ln_kernel<<<T, 256, 0, stream>>>(x1, ln2_g, ln2_b, lnB);
    // FC1 + gelu -> bf16
    gemm_bf16_k<1><<<dim3(FF / 128, T / 256), 256, 0, stream>>>(
        lnB, wfc1B, b_fc1, nullptr, (void*)fc1B, T, FF, Dm);
    // FC2 + residual -> f32 d_out
    gemm_bf16_k<2><<<dim3(Dm / 128, T / 256), 256, 0, stream>>>(
        fc1B, wfc2B, b_fc2, x1, d_out, T, Dm, FF);
}
